// GATLayer_26560077758614
// MI455X (gfx1250) — matrix operand order repaired and checked by reading, not compiled, not run on hardware
//
#include <hip/hip_runtime.h>
#include <hip/hip_bf16.h>
#include <math.h>

// ---------------------------------------------------------------------------
// GAT layer, fused for MI455X (gfx1250, wave32, WMMA bf16 16x16x32)
//
// Roofline: adj stream (256MB, read once) is the HBM floor (~11us @ 23.3TB/s).
// Matrix work (36.5 GFLOP) runs on v_wmma_f32_16x16x32_bf16 with fp32 accum.
// Scores are bounded in (-0.2, 1]  ->  exp never overflows  ->  single-pass
// softmax fused with the attention GEMM (row-sums accumulated in-loop).
//
// B operands are stored PRE-SWIZZLED in the ISA B-fragment layout so every
// lane loads its whole v16bf fragment as two contiguous global_load_b128.
//   element (k, n) -> chunk kc=k>>5, khalf=(k>>3)&1, i=(k&7)+8*((k>>4)&1), ntile=n>>4,
//                     lane=(n&15)+16*khalf
//   offset = ((kc*NTILES + ntile)*32 + lane)*16 + i
//
// gat kernel: 4 waves x 4 column-tiles each (vs 8x2) -> A-fragment (adj/exp)
// redundancy is 4x instead of 8x, keeping the trans-pipe work (~3us) hidden
// under the adj HBM stream.
// ---------------------------------------------------------------------------

typedef __attribute__((ext_vector_type(16))) __bf16 v16bf;
typedef __attribute__((ext_vector_type(8)))  float  v8f;

#define N_NODES 8192
#define D_IN    512
#define D_OUT   256
#define NTILES  (D_OUT / 16)
#define SLOPE   0.2f

union BFrag { v16bf v; uint4 q[2]; };

__device__ __forceinline__ v8f wmma_bf16(v16bf a, v16bf b, v8f c) {
  return __builtin_amdgcn_wmma_f32_16x16x32_bf16(
      /*neg_a=*/false, a, /*neg_b=*/false, b,
      /*c_mod=*/(short)0, c, /*reuse_a=*/false, /*reuse_b=*/false);
}

// Offset (in elements) of element (k, n) in the swizzled B layout.
__device__ __forceinline__ size_t swz_off(int k, int n) {
  const int kc    = k >> 5;
  const int khalf = (k >> 3) & 1;
  const int i     = (k & 7) | (((k >> 4) & 1) << 3);
  const int ntile = n >> 4;
  const int lane  = (n & 15) + (khalf << 4);
  return ((size_t)(kc * NTILES + ntile) * 32 + lane) * 16 + i;
}

// ------------------------------ Phase 1 ------------------------------------
// input fp32 -> bf16 (vectorized); W fp32 -> bf16 scattered into B-frag order
__global__ void cvt_kernel(const float* __restrict__ inp,
                           const float* __restrict__ W,
                           __bf16* __restrict__ inb,
                           __bf16* __restrict__ Wswz) {
  const int j = blockIdx.x * 256 + threadIdx.x;
  if (j < (N_NODES * D_IN) / 4) {
    const float4 f = ((const float4*)inp)[j];
    __bf16* o = inb + 4 * (size_t)j;
    o[0] = (__bf16)f.x; o[1] = (__bf16)f.y;
    o[2] = (__bf16)f.z; o[3] = (__bf16)f.w;
  }
  if (j < D_IN * D_OUT) {
    const int k = j >> 8;        // row of W (D_OUT = 256)
    const int n = j & 255;
    Wswz[swz_off(k, n)] = (__bf16)W[j];
  }
}

// ------------------------------ Phase 2 ------------------------------------
// h = in_bf16 @ W + b. grid: 512 stripes x 256 threads (8 waves).
// Wave w -> columns [32w, 32w+32). Writes Hf (fp32) and Hb in swizzled
// B-fragment order for the attention GEMM.
__global__ void gemm1_kernel(const __bf16* __restrict__ A,     // [8192,512]
                             const __bf16* __restrict__ Bswz,  // swizzled W
                             const float*  __restrict__ bias,
                             float*  __restrict__ Hf,          // [8192,256]
                             __bf16* __restrict__ HbSwz) {     // swizzled h
  const int stripe = blockIdx.x;
  const int wave   = threadIdx.x >> 5;
  const int lane   = threadIdx.x & 31;
  const int mloc   = lane & 15;
  const int khalf  = lane >> 4;
  const int mrow   = stripe * 16 + mloc;
  const int ntile0 = wave * 2;
  const int ntile1 = wave * 2 + 1;
  const int n0     = ntile0 * 16 + mloc;
  const int n1     = n0 + 16;

  v8f acc0 = {0.f,0.f,0.f,0.f,0.f,0.f,0.f,0.f};
  v8f acc1 = acc0;

  for (int kb = 0; kb < D_IN; kb += 32) {
    // A fragment: two contiguous 8x bf16 runs -> two b128 loads
    BFrag af;
    const uint4* ap = (const uint4*)(A + (size_t)mrow * D_IN + kb + khalf * 8);
    af.q[0] = ap[0];     // K = kb + khalf*8      .. +7
    af.q[1] = ap[2];     // K = kb + 16 + khalf*8 .. +7   (+32 bytes)

    // B fragments: 32 contiguous bytes per lane from swizzled W
    const int kc = kb >> 5;
    BFrag b0, b1;
    const uint4* p0 = (const uint4*)(Bswz + ((size_t)(kc * NTILES + ntile0) * 32 + lane) * 16);
    const uint4* p1 = (const uint4*)(Bswz + ((size_t)(kc * NTILES + ntile1) * 32 + lane) * 16);
    b0.q[0] = p0[0]; b0.q[1] = p0[1];
    b1.q[0] = p1[0]; b1.q[1] = p1[1];

    acc0 = wmma_bf16(af.v, b0.v, acc0);
    acc1 = wmma_bf16(af.v, b1.v, acc1);
  }

  const float bias0 = bias[n0];
  const float bias1 = bias[n1];
#pragma unroll
  for (int r = 0; r < 8; ++r) {
    const int m = stripe * 16 + r + 8 * khalf;   // D layout: M = r + 8*lanehalf
    const float v0 = acc0[r] + bias0;
    const float v1 = acc1[r] + bias1;
    Hf[(size_t)m * D_OUT + n0] = v0;
    Hf[(size_t)m * D_OUT + n1] = v1;
    HbSwz[swz_off(m, n0)] = (__bf16)v0;          // k-index of GEMM2 == m
    HbSwz[swz_off(m, n1)] = (__bf16)v1;
  }
}

// ------------------------------ Phase 3 ------------------------------------
__global__ void st_kernel(const float* __restrict__ Hf,
                          const float* __restrict__ a_self,
                          const float* __restrict__ a_neighs,
                          float* __restrict__ s, float* __restrict__ t) {
  const int row  = blockIdx.x * 8 + (threadIdx.x >> 5);
  const int lane = threadIdx.x & 31;
  float ss = 0.f, tt = 0.f;
  const float* hr = Hf + (size_t)row * D_OUT;
#pragma unroll
  for (int k = lane; k < D_OUT; k += 32) {
    const float hv = hr[k];
    ss += hv * a_self[k];
    tt += hv * a_neighs[k];
  }
#pragma unroll
  for (int off = 16; off > 0; off >>= 1) {
    ss += __shfl_down(ss, off, 32);
    tt += __shfl_down(tt, off, 32);
  }
  if (lane == 0) { s[row] = ss; t[row] = tt; }
}

// ------------------------------ Phase 4 ------------------------------------
// Fused masked-softmax attention + output GEMM + ELU.
// grid: 512 stripes x 128 threads (4 waves); wave w -> columns [64w, 64w+64)
// as four 16x16 tiles sharing one A fragment.
__global__ void gat_kernel(const float*  __restrict__ adj,     // [8192,8192]
                           const __bf16* __restrict__ HbSwz,   // swizzled h
                           const float*  __restrict__ s,
                           const float*  __restrict__ t,
                           const float*  __restrict__ beta_p,
                           float* __restrict__ out) {          // [8192,256]
  const int stripe = blockIdx.x;
  const int wave   = threadIdx.x >> 5;
  const int lane   = threadIdx.x & 31;
  const int mloc   = lane & 15;
  const int khalf  = lane >> 4;
  const int row    = stripe * 16 + mloc;
  const int nt     = wave * 4;                  // first of 4 column tiles

  const float beta = beta_p[0];
  const float sv   = s[row];
  const float asv  = fabsf(sv);
  const float* adjrow = adj + (size_t)row * N_NODES;

  v8f acc0 = {0.f,0.f,0.f,0.f,0.f,0.f,0.f,0.f};
  v8f acc1 = acc0, acc2 = acc0, acc3 = acc0;
  float psum = 0.f;                             // softmax denominator (partial)

#pragma unroll 2
  for (int kb = 0; kb < N_NODES; kb += 32) {
    // Prefetch upcoming adj cachelines (the HBM-bound stream)
    if (kb + 256 < N_NODES)
      __builtin_prefetch(adjrow + kb + 256, 0, 0);

    const int k0 = kb + khalf * 8;              // first 8-run of this lane
    const int k1 = kb + 16 + khalf * 8;         // second 8-run

    // Vector loads: t and adj values for this lane's 16 A elements
    float tv[16], av[16];
    *(float4*)&tv[0]  = *(const float4*)(t + k0);
    *(float4*)&tv[4]  = *(const float4*)(t + k0 + 4);
    *(float4*)&tv[8]  = *(const float4*)(t + k1);
    *(float4*)&tv[12] = *(const float4*)(t + k1 + 4);
    *(float4*)&av[0]  = *(const float4*)(adjrow + k0);
    *(float4*)&av[4]  = *(const float4*)(adjrow + k0 + 4);
    *(float4*)&av[8]  = *(const float4*)(adjrow + k1);
    *(float4*)&av[12] = *(const float4*)(adjrow + k1 + 4);

    // Build A fragment P[16x32]: masked exp scores -> bf16
    BFrag af;
#pragma unroll
    for (int i = 0; i < 16; ++i) {
      const float cosv = (sv * tv[i]) / (asv * fabsf(tv[i]) + 1e-7f);
      const float lk   = cosv > 0.f ? cosv : SLOPE * cosv;
      const float w    = av[i] > 0.f ? __expf(beta * lk) : 0.f;
      psum += w;
      af.v[i] = (__bf16)w;
    }

    // B fragments: 32 contiguous bytes/lane from swizzled h (L2-resident)
    const size_t base = ((size_t)((kb >> 5) * NTILES + nt) * 32 + lane) * 16;
    BFrag b0, b1, b2, b3;
    const uint4* p0 = (const uint4*)(HbSwz + base);
    const uint4* p1 = (const uint4*)(HbSwz + base + 512);    // +1 ntile
    const uint4* p2 = (const uint4*)(HbSwz + base + 1024);
    const uint4* p3 = (const uint4*)(HbSwz + base + 1536);
    b0.q[0] = p0[0]; b0.q[1] = p0[1];
    b1.q[0] = p1[0]; b1.q[1] = p1[1];
    b2.q[0] = p2[0]; b2.q[1] = p2[1];
    b3.q[0] = p3[0]; b3.q[1] = p3[1];

    acc0 = wmma_bf16(af.v, b0.v, acc0);
    acc1 = wmma_bf16(af.v, b1.v, acc1);
    acc2 = wmma_bf16(af.v, b2.v, acc2);
    acc3 = wmma_bf16(af.v, b3.v, acc3);
  }

  // Row sums: lanes m and m^16 hold complementary K halves of row m
  psum += __shfl_xor(psum, 16, 32);             // lane l: rowsum of row (l&15)

#pragma unroll
  for (int r = 0; r < 8; ++r) {
    const int m   = r + 8 * khalf;              // D-layout row for VGPR r
    const float rs  = __shfl(psum, m, 32);
    const float inv = 1.0f / rs;
    const int grow  = stripe * 16 + m;
    float* orow = out + (size_t)grow * D_OUT + nt * 16 + mloc;
    float v0 = acc0[r] * inv;
    float v1 = acc1[r] * inv;
    float v2 = acc2[r] * inv;
    float v3 = acc3[r] * inv;
    v0 = v0 > 0.f ? v0 : (__expf(v0) - 1.0f);   // ELU
    v1 = v1 > 0.f ? v1 : (__expf(v1) - 1.0f);
    v2 = v2 > 0.f ? v2 : (__expf(v2) - 1.0f);
    v3 = v3 > 0.f ? v3 : (__expf(v3) - 1.0f);
    orow[0]  = v0;
    orow[16] = v1;
    orow[32] = v2;
    orow[48] = v3;
  }
}

// ------------------------------ launcher -----------------------------------
extern "C" void kernel_launch(void* const* d_in, const int* in_sizes, int n_in,
                              void* d_out, int out_size, void* d_ws, size_t ws_size,
                              hipStream_t stream) {
  const float* inp      = (const float*)d_in[0];   // [8192,512]
  const float* adj      = (const float*)d_in[1];   // [8192,8192]
  // d_in[2] = M (unused by forward)
  const float* W        = (const float*)d_in[3];   // [512,256]
  const float* b        = (const float*)d_in[4];   // [256]
  const float* a_self   = (const float*)d_in[5];   // [256]
  const float* a_neighs = (const float*)d_in[6];   // [256]
  const float* beta     = (const float*)d_in[7];   // [1]
  float* out = (float*)d_out;

  char* ws = (char*)d_ws;
  __bf16* inb   = (__bf16*)(ws);                   //  8,388,608 B
  __bf16* Wswz  = (__bf16*)(ws + 8388608);         //    262,144 B
  float*  Hf    = (float* )(ws + 8650752);         //  8,388,608 B
  __bf16* HbSwz = (__bf16*)(ws + 17039360);        //  4,194,304 B
  float*  sP    = (float* )(ws + 21233664);        //     32,768 B
  float*  tP    = (float* )(ws + 21266432);        //     32,768 B

  cvt_kernel  <<<(N_NODES * D_IN / 4 + 255) / 256, 256, 0, stream>>>(inp, W, inb, Wswz);
  gemm1_kernel<<<N_NODES / 16, 256, 0, stream>>>(inb, Wswz, b, Hf, HbSwz);
  st_kernel   <<<N_NODES / 8, 256, 0, stream>>>(Hf, a_self, a_neighs, sP, tP);
  gat_kernel  <<<N_NODES / 16, 128, 0, stream>>>(adj, HbSwz, sP, tP, beta, out);
}
